// TriAttention_25305947308319
// MI455X (gfx1250) — matrix operand order repaired and checked by reading, not compiled, not run on hardware
//
#include <hip/hip_runtime.h>

typedef __attribute__((ext_vector_type(16))) _Float16 v16h;
typedef __attribute__((ext_vector_type(8)))  _Float16 h8;
typedef __attribute__((ext_vector_type(4)))  _Float16 h4;
typedef __attribute__((ext_vector_type(2)))  _Float16 h2;
typedef __attribute__((ext_vector_type(8)))  float    v8f;

extern "C" __device__ _Float16 __ocml_exp2_f16(_Float16);

#define S_LEN 256
#define HEADS 4
#define DDIM  32

// Padded LDS row strides (halves). Rows land 4 LDS banks apart so 16
// consecutive rows tile all 64 banks -> conflict-free ds_load_b128 frags.
#define QK_STRIDE 40    // 80 B  = 20 dwords; r*20 mod 64: spacing 4, 16 distinct
#define VT_STRIDE 264   // 528 B = 132 dwords ≡ 4 (mod 64)
#define SW_STRIDE 264

union V16 { v16h v; h2 p[8]; };

// ---- WMMA fragment loaders (CDNA5 ISA 7.12.2 layouts, wave32) ----
// A-matrix 16x32 f16: lane<16 -> row=lane,    K halves {0..7, 16..23};
//                     lane>=16 -> row=lane-16, K halves {8..15, 24..31}
__device__ __forceinline__ v16h load_afrag(const _Float16* t, int rstride, int lane) {
  const int row = lane & 15;
  const int kb  = (lane & 16) ? 8 : 0;
  const _Float16* p = t + row * rstride + kb;
  union { v16h v; struct { h8 lo, hi; } s; } u;
  u.s.lo = *(const h8*)(p);
  u.s.hi = *(const h8*)(p + 16);
  return u.v;
}

// B-matrix 32x16 f16: lane -> column (lane&15); same K order as A:
// lanes<16 hold K {0..7, 16..23}, lanes>=16 hold K {8..15, 24..31}.
// LDS tile stored as [N=16 rows][K cols] row-major (i.e. B transposed).
__device__ __forceinline__ v16h load_bfrag(const _Float16* t, int rstride, int lane) {
  const int row = lane & 15;
  const int kb  = (lane & 16) ? 8 : 0;
  const _Float16* p = t + row * rstride + kb;
  union { v16h v; struct { h8 lo, hi; } s; } u;
  u.s.lo = *(const h8*)(p);
  u.s.hi = *(const h8*)(p + 16);
  return u.v;
}

__global__ __launch_bounds__(256) void tri_attn_wmma_kernel(
    const float* __restrict__ Q, const float* __restrict__ K,
    const float* __restrict__ V, const float* __restrict__ B1,
    const float* __restrict__ B2, float* __restrict__ O)
{
  // LDS: Q,K f16 [i][d] (20KB each, padded); V transposed f16 [d][j] (16.5KB);
  //      per-wave logit tile 16x264 f16 (66KB total). ~124KB -> 2 blocks/WGP.
  __shared__ _Float16 Qh[S_LEN * QK_STRIDE];
  __shared__ _Float16 Kh[S_LEN * QK_STRIDE];
  __shared__ _Float16 Vt[DDIM * VT_STRIDE];
  __shared__ _Float16 Sw[8 * 16 * SW_STRIDE];

  const int tid  = threadIdx.x;
  const int lane = tid & 31;
  const int wave = tid >> 5;
  const int n = blockIdx.x >> 2;   // / HEADS
  const int h = blockIdx.x & 3;    // % HEADS

  // (b,n,i,h,d) flat: ((n*256 + i)*4 + h)*32 + d  -> row stride 128 floats
  const float* Qg = Q + (size_t)n * (S_LEN * HEADS * DDIM) + h * DDIM;
  const float* Kg = K + (size_t)n * (S_LEN * HEADS * DDIM) + h * DDIM;
  const float* Vg = V + (size_t)n * (S_LEN * HEADS * DDIM) + h * DDIM;
  const float* b1 = B1 + (size_t)n * S_LEN;              // (B,N,1,1,S)
  const float* b2 = B2 + (size_t)h * (S_LEN * S_LEN);    // (B,1,H,S,S)
  float*       Og = O + (size_t)n * (S_LEN * HEADS * DDIM) + h * DDIM;

  // ---- Stage Q,K -> LDS f16 [i][d] (coalesced float4 reads) ----
  for (int e = tid; e < S_LEN * 8; e += 256) {   // 2048 float4 per tensor
    const int i = e >> 3, f = e & 7;
    const float4 q4 = *(const float4*)(Qg + i * 128 + f * 4);
    const float4 k4 = *(const float4*)(Kg + i * 128 + f * 4);
    h4 qh = { (_Float16)q4.x, (_Float16)q4.y, (_Float16)q4.z, (_Float16)q4.w };
    h4 kh = { (_Float16)k4.x, (_Float16)k4.y, (_Float16)k4.z, (_Float16)k4.w };
    *(h4*)(Qh + i * QK_STRIDE + f * 4) = qh;
    *(h4*)(Kh + i * QK_STRIDE + f * 4) = kh;
  }
  // ---- Stage V -> LDS f16 transposed [d][j] ----
  for (int e = tid; e < S_LEN * DDIM; e += 256) {
    const int j = e >> 5, d = e & 31;
    Vt[d * VT_STRIDE + j] = (_Float16)Vg[j * 128 + d];
  }
  __syncthreads();

  // Logits are stored pre-scaled by log2(e): s' = log2e*(qk/sqrt(D) + b1 + b2)
  const float C0    = 0.25506953149031775f;  // log2(e)/sqrt(32)
  const float LOG2E = 1.4426950408889634f;

  _Float16* Sm = Sw + wave * (16 * SW_STRIDE);  // this wave's logit tile
  const int rowoff = (lane & 16) ? 8 : 0;       // C-layout row offset per half-wave
  const int cl = lane & 15;                     // C-layout column

  v16h onesf;                                   // all-ones B matrix for row sums
#pragma unroll
  for (int i = 0; i < 16; ++i) onesf[i] = (_Float16)1.0f;

  // Each wave: 2 query tiles of 16 rows (tiles wave and wave+8)
  for (int t = 0; t < 2; ++t) {
    const int i0 = (wave + t * 8) * 16;
    const v16h qf = load_afrag(Qh + i0 * QK_STRIDE, QK_STRIDE, lane);

    // ---- Phase 1: S' = QK^T*c0 + log2e*(b1+b2) -> LDS; re-read as A-frags;
    //      row-max via packed f16 max (rows live in lanes in A layout)
    V16 sfr[8];
    h2 mx2 = { (_Float16)-60000.0f, (_Float16)-60000.0f };
#pragma unroll
    for (int jc = 0; jc < 8; ++jc) {
      const int j0 = jc * 32;
      const v16h kf0 = load_bfrag(Kh + j0 * QK_STRIDE, QK_STRIDE, lane);
      const v16h kf1 = load_bfrag(Kh + (j0 + 16) * QK_STRIDE, QK_STRIDE, lane);
      const v8f z = {};
      v8f s0 = __builtin_amdgcn_wmma_f32_16x16x32_f16(false, qf, false, kf0,
                                                      (short)0, z, false, false);
      v8f s1 = __builtin_amdgcn_wmma_f32_16x16x32_f16(false, qf, false, kf1,
                                                      (short)0, z, false, false);
      // Batch all bias loads for this chunk before any arithmetic (one wait)
      const float* b2p = b2 + (size_t)(i0 + rowoff) * S_LEN + j0 + cl;
      float b2v[16];
#pragma unroll
      for (int r = 0; r < 8; ++r) {
        b2v[r]     = b2p[r * S_LEN];
        b2v[8 + r] = b2p[r * S_LEN + 16];
      }
      const float b1a = b1[j0 + cl]      * LOG2E;
      const float b1b = b1[j0 + 16 + cl] * LOG2E;
#pragma unroll
      for (int r = 0; r < 8; ++r) {
        const float sa = fmaf(s0[r], C0, fmaf(b2v[r],     LOG2E, b1a));
        const float sb = fmaf(s1[r], C0, fmaf(b2v[8 + r], LOG2E, b1b));
        Sm[(r + rowoff) * SW_STRIDE + j0 + cl]      = (_Float16)sa;
        Sm[(r + rowoff) * SW_STRIDE + j0 + 16 + cl] = (_Float16)sb;
      }
      // read back this chunk in A layout (same wave: LDS ops are in-order)
      sfr[jc].v = load_afrag(Sm + j0, SW_STRIDE, lane);
#pragma unroll
      for (int i = 0; i < 8; ++i)
        mx2 = __builtin_elementwise_max(mx2, sfr[jc].p[i]);
    }
    // fold packed halves, then combine the two half-wave key subsets
    float mf = (float)__builtin_elementwise_max(mx2[0], mx2[1]);
    mf = fmaxf(mf, __shfl_xor(mf, 16, 32));
    const _Float16 mh = (_Float16)mf;
    const h2 mxb = { mh, mh };

    // ---- Phase 2: p = exp2(s' - m') fully in f16 (already A layout).
    //      PV WMMAs + a third WMMA vs all-ones B gives row sums in C layout.
    v8f o_lo = {}; v8f o_hi = {}; v8f o_sum = {};
#pragma unroll
    for (int c = 0; c < 8; ++c) {
      const int j0 = c * 32;
      V16 pf;
#pragma unroll
      for (int i = 0; i < 8; ++i) {
        const h2 d = sfr[c].p[i] - mxb;            // v_pk_add_f16 (neg)
        h2 e;
        e[0] = __ocml_exp2_f16(d[0]);              // v_exp_f16
        e[1] = __ocml_exp2_f16(d[1]);
        pf.p[i] = e;
      }
      const v16h vf0 = load_bfrag(Vt + j0, VT_STRIDE, lane);                  // d 0..15
      const v16h vf1 = load_bfrag(Vt + 16 * VT_STRIDE + j0, VT_STRIDE, lane); // d 16..31
      o_lo  = __builtin_amdgcn_wmma_f32_16x16x32_f16(false, pf.v, false, vf0,
                                                     (short)0, o_lo,  false, false);
      o_hi  = __builtin_amdgcn_wmma_f32_16x16x32_f16(false, pf.v, false, vf1,
                                                     (short)0, o_hi,  false, false);
      o_sum = __builtin_amdgcn_wmma_f32_16x16x32_f16(false, pf.v, false, onesf,
                                                     (short)0, o_sum, false, false);
    }

    // ---- Normalize and store: row = i0+rowoff+r, d = cl / cl+16 ----
    //      o_sum[r] holds the row sum in the same C-layout slot as o_lo/o_hi.
#pragma unroll
    for (int r = 0; r < 8; ++r) {
      const float inv = 1.0f / o_sum[r];
      const int row = i0 + rowoff + r;
      Og[row * 128 + cl]      = o_lo[r] * inv;
      Og[row * 128 + cl + 16] = o_hi[r] * inv;
    }
  }
}

extern "C" void kernel_launch(void* const* d_in, const int* in_sizes, int n_in,
                              void* d_out, int out_size, void* d_ws, size_t ws_size,
                              hipStream_t stream) {
  const float* Q  = (const float*)d_in[0];
  const float* K  = (const float*)d_in[1];
  const float* V  = (const float*)d_in[2];
  const float* B1 = (const float*)d_in[3];
  const float* B2 = (const float*)d_in[4];
  float* O = (float*)d_out;
  (void)in_sizes; (void)n_in; (void)out_size; (void)d_ws; (void)ws_size;

  dim3 grid(S_LEN * HEADS);   // one block per (n, h) problem: 1024 blocks
  dim3 block(256);            // 8 waves of 32
  tri_attn_wmma_kernel<<<grid, block, 0, stream>>>(Q, K, V, B1, B2, O);
}
